// LinearSchNet_49143015801197
// MI455X (gfx1250) — matrix operand order repaired and checked by reading, not compiled, not run on hardware
//
#include <hip/hip_runtime.h>
#include <hip/hip_bf16.h>
#include <math.h>

// ---------------------------------------------------------------------------
// SchNet forward for MI455X (gfx1250, wave32, WMMA + Tensor Data Mover).
// Dense GEMMs -> v_wmma_f32_16x16x32_bf16 (f32 accumulate).
// Per-layer weights staged into LDS once per 8-wave workgroup via TDM
// (tensor_load_to_lds / s_wait_tensorcnt); B-fragments then come from LDS.
// Filter-MLP -> cutoff -> gather -> scatter-add fused (Wf never hits HBM).
// ---------------------------------------------------------------------------

typedef __bf16 bf16;
typedef __attribute__((ext_vector_type(16))) __bf16 v16bf;
typedef __attribute__((ext_vector_type(8)))  float  v8f;
typedef unsigned int u32;
typedef __attribute__((ext_vector_type(4))) unsigned int u32x4;
typedef __attribute__((ext_vector_type(4))) int i32x4;
typedef __attribute__((ext_vector_type(8))) int i32x8;

#define N_ATOMS  10000
#define E_EDGES  160000
#define EP_EDGES 80000
#define H_DIM    128
#define G_DIM    50
#define L_LAYERS 6
#define CUTOFF   10.0f
#define LN2F     0.69314718056f

union FragU { v16bf v; u32 u[8]; };
union PackU { bf16 b[2]; u32 u; };

__device__ inline float d_ssp(float x) {            // softplus(x) - ln2, stable
  float ax = fabsf(x);
  return fmaxf(x, 0.0f) + __logf(1.0f + __expf(-ax)) - LN2F;
}
__device__ inline float d_sigmoid(float x) { return 1.0f / (1.0f + __expf(-x)); }

__device__ inline v8f splat8(float x) {
  v8f c;
#pragma unroll
  for (int i = 0; i < 8; ++i) c[i] = x;
  return c;
}

__device__ inline v8f wmma_bf16(v16bf a, v16bf b, v8f c) {
  // D = A(16x32 bf16) * B(32x16 bf16) + C(16x16 f32)
  return __builtin_amdgcn_wmma_f32_16x16x32_bf16(false, a, false, b, (short)0, c,
                                                 false, false);
}

// ----- Tensor Data Mover: 1-D bulk load global -> LDS (ISA 08, D# groups) ---
// nbytes must be a multiple of 8 and <= 65535*8. One descriptor, 8B elements.
__device__ inline void tdm_load_1d(u32 lds_off, const void* gaddr, u32 nbytes) {
  u32 n8 = nbytes >> 3;                       // element count (data_size=8B)
  size_t ga = (size_t)gaddr;
  u32x4 g0;
  g0[0] = 1u;                                 // count=1, user descriptor
  g0[1] = lds_off;                            // D#.lds_addr (bytes)
  g0[2] = (u32)(ga & 0xFFFFFFFFu);            // global_addr[31:0]
  g0[3] = (u32)((ga >> 32) & 0x1FFFFFFu)      // global_addr[56:32]
          | (2u << 30);                       // type=2 ("image")
  i32x8 g1;
  g1[0] = (int)(3u << 16);                    // wg_mask=0, data_size=3 (8B)
  g1[1] = (int)((n8 & 0xFFFFu) << 16);        // tensor_dim0[15:0]
  g1[2] = (int)(((n8 >> 16) & 0xFFFFu)        // tensor_dim0[31:16]
          | (1u << 16));                      // tensor_dim1 = 1
  g1[3] = (int)((n8 & 0xFFFFu) << 16);        // tile_dim0 = n8 (tile_dim1=0)
  g1[4] = 0;                                  // tile_dim1=0, tile_dim2=0
  g1[5] = (int)n8;                            // tensor_dim0_stride[31:0]
  g1[6] = 0;
  g1[7] = 0;
  i32x4 z4 = {0, 0, 0, 0};
#if __clang_major__ >= 23
  i32x8 z8 = {0, 0, 0, 0, 0, 0, 0, 0};
  __builtin_amdgcn_tensor_load_to_lds(g0, g1, z4, z4, z8, 0);
#else
  __builtin_amdgcn_tensor_load_to_lds(g0, g1, z4, z4, 0);
#endif
}

// A-fragment (16x32, bf16, row-major source with leading dim ldK elements).
// Per ISA 7.12.2: lane(0-15)=row M, half=lane>>4; VGPR v holds K pair
//   K0 = 2*(v&3) + 8*half + 16*(v>>2)   (packed 2 bf16 per dword)
__device__ inline v16bf ldA_bf(const bf16* Xrow0, int ldK, int kb, int lane) {
  int hh = (lane >> 4) & 1, r = lane & 15;
  const u32* p = (const u32*)(Xrow0 + (size_t)r * ldK + kb);
  FragU f;
#pragma unroll
  for (int v = 0; v < 8; ++v)
    f.u[v] = p[(v & 3) + 4 * hh + 8 * (v >> 2)];
  return f.v;
}

// A-fragment built from an f32 source (converted to bf16 on load).
__device__ inline v16bf ldA_f32(const float* Xrow0, int ldK, int kb, int lane) {
  int hh = (lane >> 4) & 1, r = lane & 15;
  const float* p = Xrow0 + (size_t)r * ldK + kb;
  FragU f;
#pragma unroll
  for (int v = 0; v < 8; ++v) {
    int k0 = 2 * (v & 3) + 8 * hh + 16 * (v >> 2);
    PackU pk; pk.b[0] = (bf16)p[k0]; pk.b[1] = (bf16)p[k0 + 1];
    f.u[v] = pk.u;
  }
  return f.v;
}

// B-fragment (32x16, bf16). Weights pre-transposed: Wt[N][K] (K contiguous).
// lane(0-15)=col N; VGPR v holds K pair K0 = 2*(v&3) + 8*half + 16*(v>>2).
__device__ inline v16bf ldB(const bf16* Wt, int ldK, int nb, int kb, int lane) {
  int hh = (lane >> 4) & 1, r = lane & 15;
  const u32* p = (const u32*)(Wt + (size_t)(nb + r) * ldK + kb);
  FragU f;
#pragma unroll
  for (int v = 0; v < 8; ++v)
    f.u[v] = p[(v & 3) + 4 * hh + 8 * (v >> 2)];
  return f.v;
}

__device__ inline void atomAddF(float* p, float v) {
  unsafeAtomicAdd(p, v);   // global_atomic_add_f32
}

// ---------------------------------------------------------------------------
// Prep kernels
// ---------------------------------------------------------------------------

// Convert + transpose + zero-pad weight [K,Nn] f32 (row major) -> [Np,Kp] bf16.
__global__ void wprep_kernel(const float* __restrict__ src, bf16* __restrict__ dst,
                             int K, int Nn, int Kp, int Np) {
  int t = blockIdx.x * blockDim.x + threadIdx.x;
  if (t >= Kp * Np) return;
  int np = t / Kp, kp = t - np * Kp;
  float v = (kp < K && np < Nn) ? src[(size_t)kp * Nn + np] : 0.0f;
  dst[t] = (bf16)v;
}

__global__ void zero_kernel(float* __restrict__ p, int n) {
  int t = blockIdx.x * blockDim.x + threadIdx.x;
  if (t < n) p[t] = 0.0f;
}

// h = emb[z]  (f32 master copy + bf16 shadow for WMMA inputs)
__global__ void emb_kernel(const int* __restrict__ z, const float* __restrict__ emb,
                           float* __restrict__ h, bf16* __restrict__ hbf) {
  int t = blockIdx.x * blockDim.x + threadIdx.x;
  if (t >= N_ATOMS * H_DIM) return;
  int n = t >> 7, c = t & 127;
  float v = emb[(size_t)z[n] * H_DIM + c];
  h[t] = v;
  hbf[t] = (bf16)v;
}

// Per-edge geometry: distance -> Gaussian smearing (50 -> padded 64, bf16,
// packed dword stores) and cosine cutoff C.
__global__ void geom_kernel(const float* __restrict__ pos, const int* __restrict__ src,
                            const int* __restrict__ dst, bf16* __restrict__ ea,
                            float* __restrict__ Cg) {
  int e = blockIdx.x * blockDim.x + threadIdx.x;
  if (e >= E_EDGES) return;
  int s = src[e], d = dst[e];
  float dx = pos[s * 3 + 0] - pos[d * 3 + 0];
  float dy = pos[s * 3 + 1] - pos[d * 3 + 1];
  float dz = pos[s * 3 + 2] - pos[d * 3 + 2];
  float w = sqrtf(dx * dx + dy * dy + dz * dz + 1e-12f);
  const float delta = CUTOFF / (float)(G_DIM - 1);
  const float coeff = -0.5f / (delta * delta);
  u32* row = (u32*)(ea + (size_t)e * 64);
#pragma unroll
  for (int g2 = 0; g2 < 32; ++g2) {
    PackU pk;
#pragma unroll
    for (int j = 0; j < 2; ++j) {
      int g = 2 * g2 + j;
      float v = 0.0f;
      if (g < G_DIM) { float t = w - (float)g * delta; v = __expf(coeff * t * t); }
      pk.b[j] = (bf16)v;
    }
    row[g2] = pk.u;
  }
  Cg[e] = 0.5f * (__cosf(w * 3.14159265358979f / CUTOFF) + 1.0f);
}

// ---------------------------------------------------------------------------
// Interaction-layer kernels: 256 threads = 8 wave32s per block.
// Wave 0 DMAs the layer weights into LDS (TDM), all waves share them.
// Each wave owns one 16-row tile.
// ---------------------------------------------------------------------------

// xh = h_bf @ cf_lin1_w (no bias), bf16 out [N,128]
__global__ __launch_bounds__(256) void gemm_xh_kernel(
    const bf16* __restrict__ hbf, const bf16* __restrict__ lin1t,
    bf16* __restrict__ xh) {
  __shared__ __align__(16) bf16 ws[128 * 128];       // 32 KB
  const int tid = threadIdx.x;
  const int wv = tid >> 5, lane = tid & 31;
  const int hh = (lane >> 4) & 1, r = lane & 15;
  if (wv == 0) {
    tdm_load_1d((u32)(size_t)ws, lin1t, 128 * 128 * 2);
    __builtin_amdgcn_s_wait_tensorcnt(0);
  }
  __syncthreads();
  const int row0 = blockIdx.x * 128 + wv * 16;
  if (row0 >= N_ATOMS) return;
  v16bf a[4];
#pragma unroll
  for (int k = 0; k < 4; ++k) a[k] = ldA_bf(hbf + (size_t)row0 * 128, 128, k * 32, lane);
#pragma unroll
  for (int nt = 0; nt < 8; ++nt) {
    int nb = nt * 16;
    v8f c = splat8(0.0f);
#pragma unroll
    for (int k = 0; k < 4; ++k)
      c = wmma_bf16(a[k], ldB(ws, 128, nb, k * 32, lane), c);
#pragma unroll
    for (int i = 0; i < 8; ++i)
      xh[(size_t)(row0 + i + 8 * hh) * 128 + nb + r] = (bf16)c[i];
  }
}

// Fused: T = ssp(ea @ W1 + b1) ; Wf = (T @ W2 + b2) * C ;
//        agg[dst[e]] += xh[src[e]] * Wf[e]   (per column, f32 atomics)
__global__ __launch_bounds__(256) void filter_scatter_kernel(
    const bf16* __restrict__ ea, const float* __restrict__ Cg,
    const bf16* __restrict__ w1t, const float* __restrict__ b1,
    const bf16* __restrict__ w2t, const float* __restrict__ b2,
    const bf16* __restrict__ xh, const int* __restrict__ src,
    const int* __restrict__ dst, float* __restrict__ agg) {
  __shared__ __align__(16) bf16 w1s[128 * 64];       // 16 KB
  __shared__ __align__(16) bf16 w2s[128 * 128];      // 32 KB
  __shared__ __align__(16) bf16 tiles[8 * 16 * 128]; // 32 KB (per-wave slices)
  const int tid = threadIdx.x;
  const int wv = tid >> 5, lane = tid & 31;
  const int hh = (lane >> 4) & 1, r = lane & 15;

  if (wv == 0) {
    tdm_load_1d((u32)(size_t)w1s, w1t, 128 * 64 * 2);
    tdm_load_1d((u32)(size_t)w2s, w2t, 128 * 128 * 2);
    __builtin_amdgcn_s_wait_tensorcnt(0);
  }
  __syncthreads();

  const int row0 = blockIdx.x * 128 + wv * 16;       // E % 128 == 0
  bf16* tile = tiles + wv * (16 * 128);

  // Stage 1: [16,64] @ [64,128] -> ssp -> private LDS tile (bf16)
  v16bf a0 = ldA_bf(ea + (size_t)row0 * 64, 64, 0, lane);
  v16bf a1 = ldA_bf(ea + (size_t)row0 * 64, 64, 32, lane);
#pragma unroll
  for (int nt = 0; nt < 8; ++nt) {
    int nb = nt * 16;
    v8f c = splat8(b1[nb + r]);
    c = wmma_bf16(a0, ldB(w1s, 64, nb, 0, lane), c);
    c = wmma_bf16(a1, ldB(w1s, 64, nb, 32, lane), c);
#pragma unroll
    for (int i = 0; i < 8; ++i)
      tile[(i + 8 * hh) * 128 + nb + r] = (bf16)d_ssp(c[i]);
  }

  // Edge metadata for the 16 rows this wave owns
  float ce[8]; int se[8], de[8];
#pragma unroll
  for (int i = 0; i < 8; ++i) {
    int e = row0 + i + 8 * hh;
    ce[i] = Cg[e]; se[i] = src[e]; de[i] = dst[e];
  }
  v16bf ta[4];                                        // same-wave LDS RAW: DScnt
#pragma unroll
  for (int k = 0; k < 4; ++k) ta[k] = ldA_bf(tile, 128, k * 32, lane);

  // Stage 2: [16,128] @ [128,128], then fused cutoff * gather * scatter-add
#pragma unroll
  for (int nt = 0; nt < 8; ++nt) {
    int nb = nt * 16;
    v8f c = splat8(b2[nb + r]);
#pragma unroll
    for (int k = 0; k < 4; ++k)
      c = wmma_bf16(ta[k], ldB(w2s, 128, nb, k * 32, lane), c);
    int col = nb + r;
#pragma unroll
    for (int i = 0; i < 8; ++i) {
      float wf = c[i] * ce[i];
      float xv = (float)xh[(size_t)se[i] * 128 + col];
      atomAddF(&agg[(size_t)de[i] * 128 + col], xv * wf);
    }
  }
}

// Fused node update: v = ssp(agg @ cf_lin2 + cb) @ lin_w + lb ; h += v
__global__ __launch_bounds__(256) void node_update_kernel(
    const float* __restrict__ agg, const bf16* __restrict__ lin2t,
    const float* __restrict__ cb, const bf16* __restrict__ lint,
    const float* __restrict__ lb, float* __restrict__ h, bf16* __restrict__ hbf) {
  __shared__ __align__(16) bf16 l2s[128 * 128];      // 32 KB
  __shared__ __align__(16) bf16 lns[128 * 128];      // 32 KB
  __shared__ __align__(16) bf16 tiles[8 * 16 * 128]; // 32 KB
  const int tid = threadIdx.x;
  const int wv = tid >> 5, lane = tid & 31;
  const int hh = (lane >> 4) & 1, r = lane & 15;

  if (wv == 0) {
    tdm_load_1d((u32)(size_t)l2s, lin2t, 128 * 128 * 2);
    tdm_load_1d((u32)(size_t)lns, lint, 128 * 128 * 2);
    __builtin_amdgcn_s_wait_tensorcnt(0);
  }
  __syncthreads();

  const int row0 = blockIdx.x * 128 + wv * 16;
  if (row0 >= N_ATOMS) return;
  bf16* tile = tiles + wv * (16 * 128);

  v16bf a[4];
#pragma unroll
  for (int k = 0; k < 4; ++k)
    a[k] = ldA_f32(agg + (size_t)row0 * 128, 128, k * 32, lane);
#pragma unroll
  for (int nt = 0; nt < 8; ++nt) {
    int nb = nt * 16;
    v8f c = splat8(cb[nb + r]);
#pragma unroll
    for (int k = 0; k < 4; ++k)
      c = wmma_bf16(a[k], ldB(l2s, 128, nb, k * 32, lane), c);
#pragma unroll
    for (int i = 0; i < 8; ++i)
      tile[(i + 8 * hh) * 128 + nb + r] = (bf16)d_ssp(c[i]);
  }
  v16bf ta[4];
#pragma unroll
  for (int k = 0; k < 4; ++k) ta[k] = ldA_bf(tile, 128, k * 32, lane);
#pragma unroll
  for (int nt = 0; nt < 8; ++nt) {
    int nb = nt * 16;
    v8f c = splat8(lb[nb + r]);
#pragma unroll
    for (int k = 0; k < 4; ++k)
      c = wmma_bf16(ta[k], ldB(lns, 128, nb, k * 32, lane), c);
#pragma unroll
    for (int i = 0; i < 8; ++i) {
      size_t idx = (size_t)(row0 + i + 8 * hh) * 128 + nb + r;
      float nh = h[idx] + c[i];
      h[idx] = nh;
      hbf[idx] = (bf16)nh;
    }
  }
}

// ---------------------------------------------------------------------------
// Output head
// ---------------------------------------------------------------------------

// hout = h_bf @ out_w + out_b  -> [N,64] f32
__global__ __launch_bounds__(256) void out_kernel(
    const bf16* __restrict__ hbf, const bf16* __restrict__ ott,
    const float* __restrict__ ob, float* __restrict__ hout) {
  __shared__ __align__(16) bf16 os[64 * 128];        // 16 KB
  const int tid = threadIdx.x;
  const int wv = tid >> 5, lane = tid & 31;
  const int hh = (lane >> 4) & 1, r = lane & 15;
  if (wv == 0) {
    tdm_load_1d((u32)(size_t)os, ott, 64 * 128 * 2);
    __builtin_amdgcn_s_wait_tensorcnt(0);
  }
  __syncthreads();
  const int row0 = blockIdx.x * 128 + wv * 16;
  if (row0 >= N_ATOMS) return;
  v16bf a[4];
#pragma unroll
  for (int k = 0; k < 4; ++k) a[k] = ldA_bf(hbf + (size_t)row0 * 128, 128, k * 32, lane);
#pragma unroll
  for (int nt = 0; nt < 4; ++nt) {
    int nb = nt * 16;
    v8f c = splat8(ob[nb + r]);
#pragma unroll
    for (int k = 0; k < 4; ++k)
      c = wmma_bf16(a[k], ldB(os, 128, nb, k * 32, lane), c);
#pragma unroll
    for (int i = 0; i < 8; ++i)
      hout[(size_t)(row0 + i + 8 * hh) * 64 + nb + r] = c[i];
  }
}

// Build ef rows: [hout[ps] (64) | hout[pd] (64) | dist | angle | 0-pad to 160]
__global__ void ef_kernel(const float* __restrict__ pos, const int* __restrict__ ps,
                          const int* __restrict__ pd, const float* __restrict__ hout,
                          bf16* __restrict__ ef) {
  int t = blockIdx.x * blockDim.x + threadIdx.x;
  if (t >= EP_EDGES * 160) return;
  int ep = t / 160, c = t - ep * 160;
  int a = ps[ep], b = pd[ep];
  float v = 0.0f;
  if (c < 64) {
    v = hout[(size_t)a * 64 + c];
  } else if (c < 128) {
    v = hout[(size_t)b * 64 + (c - 64)];
  } else if (c == 128) {
    float dx = pos[a * 3 + 0] - pos[b * 3 + 0];
    float dy = pos[a * 3 + 1] - pos[b * 3 + 1];
    float dz = pos[a * 3 + 2] - pos[b * 3 + 2];
    v = sqrtf(dx * dx + dy * dy + dz * dz + 1e-12f);
  } else if (c == 129) {
    float ax = pos[a * 3 + 0], ay = pos[a * 3 + 1], az = pos[a * 3 + 2];
    float bx = pos[b * 3 + 0], by = pos[b * 3 + 1], bz = pos[b * 3 + 2];
    float na = sqrtf(ax * ax + ay * ay + az * az);
    float nb_ = sqrtf(bx * bx + by * by + bz * bz);
    float cs = (ax * bx + ay * by + az * bz) / (na * nb_ + 1e-5f);
    cs = fminf(0.999999f, fmaxf(-0.999999f, cs));
    v = acosf(cs);
  }
  ef[t] = (bf16)v;
}

// Fused head: out = sigmoid(ef @ e_w1 + e_b1) @ e_w2 + e_b2
// GEMM1 tiles kept in registers; sigmoid + dot(e_w2) fused; cross-lane reduce.
__global__ __launch_bounds__(32) void head_kernel(
    const bf16* __restrict__ ef, const bf16* __restrict__ ew1t,
    const float* __restrict__ b1, const float* __restrict__ w2,
    const float* __restrict__ b2, float* __restrict__ out) {
  const int lane = threadIdx.x;
  const int hh = (lane >> 4) & 1, r = lane & 15;
  const int row0 = blockIdx.x * 16;
  v16bf a[5];
#pragma unroll
  for (int k = 0; k < 5; ++k) a[k] = ldA_bf(ef + (size_t)row0 * 160, 160, k * 32, lane);
  float psum[8];
#pragma unroll
  for (int i = 0; i < 8; ++i) psum[i] = 0.0f;
  for (int nt = 0; nt < 33; ++nt) {          // 528 padded cols / 16
    int n = nt * 16 + r;
    float bias = (n < 520) ? b1[n] : 0.0f;
    float wv = (n < 520) ? w2[n] : 0.0f;     // padded cols contribute 0
    v8f c = splat8(bias);
#pragma unroll
    for (int k = 0; k < 5; ++k)
      c = wmma_bf16(a[k], ldB(ew1t, 160, nt * 16, k * 32, lane), c);
#pragma unroll
    for (int i = 0; i < 8; ++i)
      psum[i] += d_sigmoid(c[i]) * wv;
  }
  float bb = b2[0];
#pragma unroll
  for (int i = 0; i < 8; ++i) {
#pragma unroll
    for (int m = 1; m < 16; m <<= 1)
      psum[i] += __shfl_xor(psum[i], m, 32);  // reduce within each 16-lane half
  }
  if (r == 0) {
#pragma unroll
    for (int i = 0; i < 8; ++i)
      out[row0 + i + 8 * hh] = psum[i] + bb;
  }
}

// ---------------------------------------------------------------------------
// Host orchestration
// ---------------------------------------------------------------------------

extern "C" void kernel_launch(void* const* d_in, const int* in_sizes, int n_in,
                              void* d_out, int out_size, void* d_ws, size_t ws_size,
                              hipStream_t stream) {
  (void)in_sizes; (void)n_in; (void)out_size; (void)ws_size;
  const int*   z      = (const int*)d_in[0];
  const float* pos    = (const float*)d_in[1];
  const int*   ei     = (const int*)d_in[2];
  const int*   pei    = (const int*)d_in[3];
  const float* embw   = (const float*)d_in[4];
  const float* mlp_w1 = (const float*)d_in[5];
  const float* mlp_b1 = (const float*)d_in[6];
  const float* mlp_w2 = (const float*)d_in[7];
  const float* mlp_b2 = (const float*)d_in[8];
  const float* cf1w   = (const float*)d_in[9];
  const float* cf2w   = (const float*)d_in[10];
  const float* cf2b   = (const float*)d_in[11];
  const float* linw   = (const float*)d_in[12];
  const float* linb   = (const float*)d_in[13];
  const float* outw   = (const float*)d_in[14];
  const float* outb   = (const float*)d_in[15];
  const float* ew1    = (const float*)d_in[16];
  const float* eb1    = (const float*)d_in[17];
  const float* ew2    = (const float*)d_in[18];
  const float* eb2    = (const float*)d_in[19];
  float* outp = (float*)d_out;

  const int* src = ei;
  const int* dst = ei + E_EDGES;
  const int* ps  = pei;
  const int* pd  = pei + EP_EDGES;

  char* ws = (char*)d_ws;
  size_t off = 0;
  auto carve = [&](size_t bytes) -> char* {
    char* p = ws + off;
    off = (off + bytes + 255) & ~(size_t)255;
    return p;
  };
  float* hf   = (float*)carve((size_t)N_ATOMS * 128 * 4);
  bf16*  hb   = (bf16*) carve((size_t)N_ATOMS * 128 * 2);
  bf16*  eab  = (bf16*) carve((size_t)E_EDGES * 64 * 2);
  float* Cg   = (float*)carve((size_t)E_EDGES * 4);
  bf16*  xh   = (bf16*) carve((size_t)N_ATOMS * 128 * 2);
  float* agg  = (float*)carve((size_t)N_ATOMS * 128 * 4);
  float* hout = (float*)carve((size_t)N_ATOMS * 64 * 4);
  bf16*  efb  = (bf16*) carve((size_t)EP_EDGES * 160 * 2);
  bf16*  w1t  = (bf16*) carve((size_t)L_LAYERS * 128 * 64 * 2);
  bf16*  w2t  = (bf16*) carve((size_t)L_LAYERS * 128 * 128 * 2);
  bf16*  l1t  = (bf16*) carve((size_t)L_LAYERS * 128 * 128 * 2);
  bf16*  l2t  = (bf16*) carve((size_t)L_LAYERS * 128 * 128 * 2);
  bf16*  lnt  = (bf16*) carve((size_t)L_LAYERS * 128 * 128 * 2);
  bf16*  ott  = (bf16*) carve((size_t)64 * 128 * 2);
  bf16*  e1t  = (bf16*) carve((size_t)528 * 160 * 2);

  // Weight conversion (f32 -> bf16, transposed to N-major, zero-padded)
  for (int l = 0; l < L_LAYERS; ++l) {
    wprep_kernel<<<(128 * 64 + 255) / 256, 256, 0, stream>>>(
        mlp_w1 + (size_t)l * 50 * 128, w1t + (size_t)l * 128 * 64, 50, 128, 64, 128);
    wprep_kernel<<<(128 * 128 + 255) / 256, 256, 0, stream>>>(
        mlp_w2 + (size_t)l * 128 * 128, w2t + (size_t)l * 128 * 128, 128, 128, 128, 128);
    wprep_kernel<<<(128 * 128 + 255) / 256, 256, 0, stream>>>(
        cf1w + (size_t)l * 128 * 128, l1t + (size_t)l * 128 * 128, 128, 128, 128, 128);
    wprep_kernel<<<(128 * 128 + 255) / 256, 256, 0, stream>>>(
        cf2w + (size_t)l * 128 * 128, l2t + (size_t)l * 128 * 128, 128, 128, 128, 128);
    wprep_kernel<<<(128 * 128 + 255) / 256, 256, 0, stream>>>(
        linw + (size_t)l * 128 * 128, lnt + (size_t)l * 128 * 128, 128, 128, 128, 128);
  }
  wprep_kernel<<<(64 * 128 + 255) / 256, 256, 0, stream>>>(outw, ott, 128, 64, 128, 64);
  wprep_kernel<<<(528 * 160 + 255) / 256, 256, 0, stream>>>(ew1, e1t, 130, 520, 160, 528);

  // Node embedding + edge geometry
  emb_kernel<<<(N_ATOMS * 128 + 255) / 256, 256, 0, stream>>>(z, embw, hf, hb);
  geom_kernel<<<(E_EDGES + 255) / 256, 256, 0, stream>>>(pos, src, dst, eab, Cg);

  const int nodeBlocks = (N_ATOMS + 127) / 128;   // 79
  // Interaction layers
  for (int l = 0; l < L_LAYERS; ++l) {
    gemm_xh_kernel<<<nodeBlocks, 256, 0, stream>>>(
        hb, l1t + (size_t)l * 128 * 128, xh);
    zero_kernel<<<(N_ATOMS * 128 + 255) / 256, 256, 0, stream>>>(agg, N_ATOMS * 128);
    filter_scatter_kernel<<<E_EDGES / 128, 256, 0, stream>>>(
        eab, Cg, w1t + (size_t)l * 128 * 64, mlp_b1 + (size_t)l * 128,
        w2t + (size_t)l * 128 * 128, mlp_b2 + (size_t)l * 128, xh, src, dst, agg);
    node_update_kernel<<<nodeBlocks, 256, 0, stream>>>(
        agg, l2t + (size_t)l * 128 * 128, cf2b + (size_t)l * 128,
        lnt + (size_t)l * 128 * 128, linb + (size_t)l * 128, hf, hb);
  }

  // Output projection + edge-prediction head
  out_kernel<<<nodeBlocks, 256, 0, stream>>>(hb, ott, outb, hout);
  ef_kernel<<<(EP_EDGES * 160 + 255) / 256, 256, 0, stream>>>(pos, ps, pd, hout, efb);
  head_kernel<<<EP_EDGES / 16, 32, 0, stream>>>(efb, e1t, eb1, ew2, eb2, outp);
}
